// Attention_79164837199973
// MI455X (gfx1250) — matrix operand order repaired and checked by reading, not compiled, not run on hardware
//
#include <hip/hip_runtime.h>
#include <hip/hip_bf16.h>

typedef __attribute__((ext_vector_type(16))) __bf16 v16bf;
typedef __attribute__((ext_vector_type(8)))  float  v8f;

#define BB 4
#define NSEQ 1024
#define CQD 512
#define CHD 512
#define HH 8
#define DD 64

union FragBF { v16bf v; uint4 q[2]; unsigned short s[16]; };
union AccF   { v8f v; float f[8]; };

__device__ __forceinline__ unsigned short f2bf(float x) {
  union { float f; unsigned u; } a; a.f = x;
  unsigned r = a.u + 0x7FFFu + ((a.u >> 16) & 1u);  // round-to-nearest-even
  return (unsigned short)(r >> 16);
}

__global__ void convert_f32_bf16(const float* __restrict__ src,
                                 unsigned short* __restrict__ dst, int n) {
  int i = blockIdx.x * blockDim.x + threadIdx.x;
  int stride = gridDim.x * blockDim.x;
  for (; i < n; i += stride) dst[i] = f2bf(src[i]);
}

#define MODE_Q 0
#define MODE_K 1
#define MODE_V 2
#define MODE_GATE 3
#define MODE_OUT 4

// C[m,n] = sum_k A[m,k] * Bw[n,k]   (A,Bw bf16 row-major, K-major weights)
// Per-wave 16x64 tile: 4 accumulators share one A fragment per K-step of 32.
__global__ void __launch_bounds__(128)
gemm_wmma(const unsigned short* __restrict__ A,
          const unsigned short* __restrict__ Bw,
          const float* __restrict__ bias,    // [N] or null
          const float* __restrict__ aux,     // gating_bias [N] or gate [M,N]
          float* __restrict__ outF,
          unsigned short* __restrict__ outB,
          int M, int N, int K, int mode) {
  const int lane = threadIdx.x & 31;
  const int wave = threadIdx.x >> 5;
  const int half = lane >> 4;      // which 16-lane group
  const int ln   = lane & 15;
  const int m0 = blockIdx.x * 16;
  const int n0 = (blockIdx.y * 4 + wave) * 64;

  AccF acc[4] = {};
  const unsigned short* arow = A + (size_t)(m0 + ln) * K;
  const unsigned short* br[4];
  for (int j = 0; j < 4; ++j) br[j] = Bw + (size_t)(n0 + j * 16 + ln) * K;

  for (int k0 = 0; k0 < K; k0 += 32) {
    FragBF af;
    // A 16x32: lanes<16 hold K {0..7,16..23}, lanes>=16 hold {8..15,24..31}
    af.q[0] = *(const uint4*)(arow + k0 + 8 * half);
    af.q[1] = *(const uint4*)(arow + k0 + 16 + 8 * half);
    for (int j = 0; j < 4; ++j) {
      FragBF bf_;  // B 32x16: lanes<16 hold K {0..7,16..23}, lanes>=16 hold {8..15,24..31}
      bf_.q[0] = *(const uint4*)(br[j] + k0 + 8 * half);
      bf_.q[1] = *(const uint4*)(br[j] + k0 + 16 + 8 * half);
      acc[j].v = __builtin_amdgcn_wmma_f32_16x16x32_bf16(false, af.v, false, bf_.v,
                                                         (short)0, acc[j].v, false, false);
    }
  }

  for (int j = 0; j < 4; ++j) {
    for (int r = 0; r < 8; ++r) {
      int m  = m0 + r + 8 * half;   // C/D row
      int nc = n0 + j * 16 + ln;    // C/D col
      float val = acc[j].f[r];
      if (mode == MODE_Q) {
        val = (val + bias[nc]) * 0.125f;  // D^-0.5 = 1/8
        int b = m >> 10, n = m & 1023, h = nc >> 6, d = nc & 63;
        outB[((size_t)(b * HH + h) * NSEQ + n) * DD + d] = f2bf(val);
      } else if (mode == MODE_K) {
        int b = m >> 10, n = m & 1023, h = nc >> 6, d = nc & 63;
        outB[((size_t)(b * HH + h) * NSEQ + n) * DD + d] = f2bf(val);
      } else if (mode == MODE_V) {  // store transposed per head: [B,H,D,N]
        int b = m >> 10, n = m & 1023, h = nc >> 6, d = nc & 63;
        outB[((size_t)(b * HH + h) * DD + d) * NSEQ + n] = f2bf(val);
      } else if (mode == MODE_GATE) {
        float g = val + bias[nc] + aux[nc];
        outF[(size_t)m * N + nc] = 1.0f / (1.0f + __expf(-g));
      } else {  // MODE_OUT: (acc + bo) * gate
        outF[(size_t)m * N + nc] = (val + bias[nc]) * aux[(size_t)m * N + nc];
      }
    }
  }
}

// Flash attention: one wave per (b, h, 16-row q tile). attn_bias tiles are
// staged into LDS with gfx1250 async global->LDS copies, overlapped with the
// QK^T WMMAs, fenced with s_wait_asynccnt.
__global__ void __launch_bounds__(32)
flash_attn(const unsigned short* __restrict__ qg,   // [B,H,N,D] bf16 (pre-scaled)
           const unsigned short* __restrict__ kg,   // [B,H,N,D] bf16
           const unsigned short* __restrict__ vtg,  // [B,H,D,N] bf16
           const float* __restrict__ biasg,         // [B,H,N,N] f32
           unsigned short* __restrict__ og) {       // [B*N, CH] bf16
  __shared__ __align__(16) unsigned short Plds[16 * 32];
  __shared__ __align__(16) float Blds[16 * 32];
  const int lane = threadIdx.x & 31;
  const int half = lane >> 4;
  const int ln   = lane & 15;
  const int m0 = blockIdx.x * 16;
  const int h  = blockIdx.y;
  const int b  = blockIdx.z;

  const unsigned short* qbase = qg  + ((size_t)(b * HH + h) * NSEQ + m0) * DD;
  const unsigned short* kbase = kg  + (size_t)(b * HH + h) * NSEQ * DD;
  const unsigned short* vbase = vtg + (size_t)(b * HH + h) * DD * NSEQ;
  const float*          bbase = biasg + ((size_t)(b * HH + h) * NSEQ + m0) * NSEQ;

  // Q A-fragments: K dim = D = 64 -> two 16x32 frags, loaded once
  FragBF qa0, qa1;
  {
    const unsigned short* qr = qbase + (size_t)ln * DD;
    qa0.q[0] = *(const uint4*)(qr + 8 * half);
    qa0.q[1] = *(const uint4*)(qr + 16 + 8 * half);
    qa1.q[0] = *(const uint4*)(qr + 32 + 8 * half);
    qa1.q[1] = *(const uint4*)(qr + 48 + 8 * half);
  }

  const unsigned ldsb = (unsigned)(uintptr_t)(&Blds[0]);  // low 32 bits = LDS offset

  AccF oacc[4] = {};   // 4 d-tiles of 16
  float rm[8], rs[8];
  for (int r = 0; r < 8; ++r) { rm[r] = -1.0e30f; rs[r] = 0.0f; }

  for (int c = 0; c < NSEQ; c += 32) {
    // Kick off async copy of the 16x32 f32 bias tile (2 KB) into LDS:
    // 4 issues x 32 lanes x 16 B, each global row is a coalesced 128 B span.
    {
      const char* gsrc = (const char*)bbase + (size_t)c * 4;
      #pragma unroll
      for (int i = 0; i < 4; ++i) {
        int off  = i * 512 + lane * 16;  // byte offset within tile
        int row  = off >> 7;             // / 128 B per row
        int colb = off & 127;
        unsigned long long ga = (unsigned long long)(uintptr_t)(gsrc + (size_t)row * (NSEQ * 4) + colb);
        unsigned la = ldsb + (unsigned)off;
        asm volatile("global_load_async_to_lds_b128 %0, %1, off"
                     :: "v"(la), "v"(ga) : "memory");
      }
    }
    if (c + 32 < NSEQ)
      __builtin_prefetch((const char*)bbase + (size_t)(c + 32) * 4, 0, 0);

    // QK^T while the bias tile is in flight
    AccF s[2] = {};
    for (int t = 0; t < 2; ++t) {
      int cc = c + 16 * t;
      FragBF kb0, kb1;   // B = K^T: B[d, col] = k[cc+col, d]; row-major k => contiguous d
      const unsigned short* kr = kbase + (size_t)(cc + ln) * DD;
      kb0.q[0] = *(const uint4*)(kr + 8 * half);
      kb0.q[1] = *(const uint4*)(kr + 16 + 8 * half);
      kb1.q[0] = *(const uint4*)(kr + 32 + 8 * half);
      kb1.q[1] = *(const uint4*)(kr + 48 + 8 * half);
      s[t].v = __builtin_amdgcn_wmma_f32_16x16x32_bf16(false, qa0.v, false, kb0.v,
                                                       (short)0, s[t].v, false, false);
      s[t].v = __builtin_amdgcn_wmma_f32_16x16x32_bf16(false, qa1.v, false, kb1.v,
                                                       (short)0, s[t].v, false, false);
    }

    asm volatile("s_wait_asynccnt 0x0" ::: "memory");
    for (int t = 0; t < 2; ++t)
      for (int r = 0; r < 8; ++r)
        s[t].f[r] += Blds[(r + 8 * half) * 32 + 16 * t + ln];

    // online softmax; row stats replicated across each 16-lane group
    for (int r = 0; r < 8; ++r) {
      float mx = fmaxf(s[0].f[r], s[1].f[r]);
      for (int off = 8; off; off >>= 1) mx = fmaxf(mx, __shfl_xor(mx, off, 32));
      float nm   = fmaxf(rm[r], mx);
      float corr = __expf(rm[r] - nm);
      rm[r] = nm;
      float p0 = __expf(s[0].f[r] - nm);
      float p1 = __expf(s[1].f[r] - nm);
      float ps = p0 + p1;
      for (int off = 8; off; off >>= 1) ps += __shfl_xor(ps, off, 32);
      rs[r] = rs[r] * corr + ps;
      oacc[0].f[r] *= corr; oacc[1].f[r] *= corr;
      oacc[2].f[r] *= corr; oacc[3].f[r] *= corr;
      // stage P (C/D layout) into LDS row-major 16x32 for A-layout reload
      Plds[(r + 8 * half) * 32 + ln]      = f2bf(p0);
      Plds[(r + 8 * half) * 32 + 16 + ln] = f2bf(p1);
    }
    __syncthreads();   // single-wave WG: lowers to S_NOP, orders LDS use
    FragBF pf;
    {
      const unsigned short* pr = &Plds[ln * 32];
      pf.q[0] = *(const uint4*)(pr + 8 * half);
      pf.q[1] = *(const uint4*)(pr + 16 + 8 * half);
    }
    for (int dt = 0; dt < 4; ++dt) {
      FragBF vb;   // B[k=token, col=d] = vT[d, token]: contiguous in token
      const unsigned short* vr = vbase + (size_t)(dt * 16 + ln) * NSEQ + c;
      vb.q[0] = *(const uint4*)(vr + 8 * half);
      vb.q[1] = *(const uint4*)(vr + 16 + 8 * half);
      oacc[dt].v = __builtin_amdgcn_wmma_f32_16x16x32_bf16(false, pf.v, false, vb.v,
                                                           (short)0, oacc[dt].v, false, false);
    }
    __syncthreads();
  }

  for (int r = 0; r < 8; ++r) {
    float inv = 1.0f / rs[r];
    int tok = m0 + r + 8 * half;
    size_t rowo = ((size_t)b * NSEQ + tok) * CHD + (size_t)h * DD;
    for (int dt = 0; dt < 4; ++dt)
      og[rowo + dt * 16 + ln] = f2bf(oacc[dt].f[r] * inv);
  }
}

extern "C" void kernel_launch(void* const* d_in, const int* in_sizes, int n_in,
                              void* d_out, int out_size, void* d_ws, size_t ws_size,
                              hipStream_t stream) {
  (void)in_sizes; (void)n_in; (void)out_size; (void)ws_size;
  const float* qx    = (const float*)d_in[0];
  const float* abias = (const float*)d_in[1];
  const float* Wq    = (const float*)d_in[2];
  const float* bq    = (const float*)d_in[3];
  const float* Wk    = (const float*)d_in[4];
  const float* Wv    = (const float*)d_in[5];
  const float* Wo    = (const float*)d_in[6];
  const float* bo    = (const float*)d_in[7];
  const float* Wg    = (const float*)d_in[8];
  const float* bg    = (const float*)d_in[9];
  const float* gbias = (const float*)d_in[10];

  const int M = BB * NSEQ;     // 4096
  size_t off = 0;
  auto alloc = [&](size_t bytes) -> void* {
    void* p = (char*)d_ws + off;
    off += (bytes + 255) & ~(size_t)255;
    return p;
  };
  unsigned short* xb  = (unsigned short*)alloc((size_t)M * CQD * 2);
  unsigned short* wqb = (unsigned short*)alloc((size_t)CHD * CQD * 2);
  unsigned short* wkb = (unsigned short*)alloc((size_t)CHD * CQD * 2);
  unsigned short* wvb = (unsigned short*)alloc((size_t)CHD * CQD * 2);
  unsigned short* wob = (unsigned short*)alloc((size_t)CQD * CHD * 2);
  unsigned short* wgb = (unsigned short*)alloc((size_t)CQD * CQD * 2);
  unsigned short* qb  = (unsigned short*)alloc((size_t)M * CHD * 2);
  unsigned short* kb  = (unsigned short*)alloc((size_t)M * CHD * 2);
  unsigned short* vtb = (unsigned short*)alloc((size_t)M * CHD * 2);
  unsigned short* ob  = (unsigned short*)alloc((size_t)M * CHD * 2);
  float*          gate = (float*)alloc((size_t)M * CQD * 4);

  // 1) fp32 -> bf16 conversions
  int nx = M * CQD;
  convert_f32_bf16<<<(nx + 255) / 256, 256, 0, stream>>>(qx, xb, nx);
  int nw = CHD * CQD;
  convert_f32_bf16<<<(nw + 255) / 256, 256, 0, stream>>>(Wq, wqb, nw);
  convert_f32_bf16<<<(nw + 255) / 256, 256, 0, stream>>>(Wk, wkb, nw);
  convert_f32_bf16<<<(nw + 255) / 256, 256, 0, stream>>>(Wv, wvb, nw);
  convert_f32_bf16<<<(nw + 255) / 256, 256, 0, stream>>>(Wo, wob, nw);
  convert_f32_bf16<<<(nw + 255) / 256, 256, 0, stream>>>(Wg, wgb, nw);

  // 2) projections + gate (WMMA GEMMs); block = 4 waves, each wave 16x64 tile
  dim3 ggrid(M / 16, CHD / 256);
  dim3 gblk(128);
  gemm_wmma<<<ggrid, gblk, 0, stream>>>(xb, wqb, bq, nullptr, nullptr, qb,
                                        M, CHD, CQD, MODE_Q);
  gemm_wmma<<<ggrid, gblk, 0, stream>>>(xb, wkb, nullptr, nullptr, nullptr, kb,
                                        M, CHD, CQD, MODE_K);
  gemm_wmma<<<ggrid, gblk, 0, stream>>>(xb, wvb, nullptr, nullptr, nullptr, vtb,
                                        M, CHD, CQD, MODE_V);
  gemm_wmma<<<ggrid, gblk, 0, stream>>>(xb, wgb, bg, gbias, gate, nullptr,
                                        M, CQD, CQD, MODE_GATE);

  // 3) flash attention (streams attn_bias exactly once, via async->LDS)
  dim3 fgrid(NSEQ / 16, HH, BB);
  flash_attn<<<fgrid, 32, 0, stream>>>(qb, kb, vtb, abias, ob);

  // 4) output projection + gating -> d_out (fp32)
  gemm_wmma<<<ggrid, gblk, 0, stream>>>(ob, wob, bo, gate, (float*)d_out, nullptr,
                                        M, CQD, CHD, MODE_OUT);
}
